// MinLSTM_91070486544679
// MI455X (gfx1250) — matrix operand order repaired and checked by reading, not compiled, not run on hardware
//
#include <hip/hip_runtime.h>
#include <hip/hip_bf16.h>
#include <stdint.h>

// Problem constants (from reference): B=4, T=8192, D=512, H=512
#define B_   4
#define T_   8192
#define D_   512
#define H_   512
#define NTM  2048   // (B*T)/16 row tiles
#define NTK  16     // D/32 k tiles
#define NTN  32     // H/16 col tiles
#define CHK  64     // scan chunks
#define CL   128    // chunk length (CHK*CL == T)

typedef __attribute__((ext_vector_type(16))) __bf16 v16bf;
typedef __attribute__((ext_vector_type(8)))  float  v8f;

// ---------- scalar helpers ----------
__device__ __forceinline__ unsigned short f2bf(float f) {
  unsigned int u = __float_as_uint(f);
  u += 0x7FFFu + ((u >> 16) & 1u);          // round-to-nearest-even
  return (unsigned short)(u >> 16);
}
__device__ __forceinline__ float softplus_f(float z) {
  if (z > 20.f)  return z;
  if (z < -20.f) return __expf(z);
  return log1pf(__expf(z));
}
__device__ __forceinline__ float log_g_f(float x) {
  // log(relu(x)+0.5) for x>=0 else -softplus(-x)
  return (x >= 0.f) ? __logf(x + 0.5f) : -softplus_f(-x);
}
__device__ __forceinline__ float logaddexp_f(float p, float q) {
  float m = fmaxf(p, q);
  if (m == -__builtin_inff()) return -__builtin_inff();
  return m + log1pf(__expf(fminf(p, q) - m));
}

// ---------- kernel 1: pack x (f32 -> bf16 WMMA A-fragment layout) ----------
// A-fragment (16-bit, 16x32): lane L<16 -> row M=L, halves e0..7 = K0..7, e8..15 = K16..23
//                              lane L>=16 -> row M=L-16, halves = K8..15, K24..31
// Storage: xp[((tm*NTK + kt)*32 + lane)*8 + dw] as packed dwords (2 bf16 each)
__global__ void pack_x_kernel(const float* __restrict__ x, uint32_t* __restrict__ xp) {
  unsigned idx  = blockIdx.x * blockDim.x + threadIdx.x;   // < NTM*NTK*32*8
  unsigned dw   = idx & 7u;
  unsigned lane = (idx >> 3) & 31u;
  unsigned kt   = (idx >> 8) & 15u;
  unsigned tm   = idx >> 12;
  unsigned row  = lane & 15u;
  unsigned koff = (lane < 16u) ? 0u : 8u;
  unsigned h0   = dw * 2u, h1 = h0 + 1u;
  unsigned K0   = ((h0 < 8u) ? h0 : h0 + 8u) + koff;
  unsigned K1   = ((h1 < 8u) ? h1 : h1 + 8u) + koff;
  const float* src = x + (size_t)(tm * 16u + row) * D_ + kt * 32u;
  uint32_t lo = f2bf(src[K0]);
  uint32_t hi = f2bf(src[K1]);
  xp[idx] = lo | (hi << 16);
}

// ---------- kernel 2: pack W_f/W_i/W_h (f32 -> bf16 WMMA B-fragment layout) ----------
// B-fragment (16-bit, 32x16): lane L<16 -> col N=L, halves e0..7 = K0..7, e8..15 = K16..23
//                              lane L>=16 -> col N=L-16, halves = K8..15, K24..31
__global__ void pack_w_kernel(const float* __restrict__ Wf, const float* __restrict__ Wi,
                              const float* __restrict__ Wh, uint32_t* __restrict__ wp) {
  unsigned idx  = blockIdx.x * blockDim.x + threadIdx.x;   // < 3*NTN*NTK*32*8
  unsigned dw   = idx & 7u;
  unsigned lane = (idx >> 3) & 31u;
  unsigned kt   = (idx >> 8) & 15u;
  unsigned tn   = (idx >> 12) & 31u;
  unsigned gate = idx >> 17;
  unsigned col  = lane & 15u;
  unsigned kb   = (lane < 16u) ? 0u : 8u;
  unsigned K0   = ((dw * 2u < 8u) ? dw * 2u : dw * 2u + 8u) + kb;
  const float* W = (gate == 0u) ? Wf : (gate == 1u) ? Wi : Wh;
  float a = W[(size_t)(kt * 32u + K0)      * H_ + tn * 16u + col];
  float b = W[(size_t)(kt * 32u + K0 + 1u) * H_ + tn * 16u + col];
  wp[idx] = (uint32_t)f2bf(a) | ((uint32_t)f2bf(b) << 16);   // per-gate arrays contiguous
}

// ---------- kernel 3: fused WMMA GEMM (3 gates) + log-gate epilogue ----------
// One wave computes a 2x2 block of 16x16 (t,h) tiles for f, i, h simultaneously:
// 12 WMMAs per k-step fed by 2 A-frag + 6 B-frag loads (0.67 KB/WMMA).
// __launch_bounds__(256, 1): allow full per-wave VGPR budget, avoid acc spills.
__global__ void __launch_bounds__(256, 1)
gemm_gates_kernel(const uint32_t* __restrict__ xp,
                  const uint32_t* __restrict__ wp,
                  const float* __restrict__ b_f,
                  const float* __restrict__ b_i,
                  const float* __restrict__ b_h,
                  float* __restrict__ logF,
                  float* __restrict__ logV) {
  const v16bf* xa = (const v16bf*)xp;
  const v16bf* wB[3];
  wB[0] = (const v16bf*)wp;                       // each gate: NTN*NTK*32 fragments
  wB[1] = wB[0] + (size_t)NTN * NTK * 32;
  wB[2] = wB[1] + (size_t)NTN * NTK * 32;

  unsigned wave = blockIdx.x * (blockDim.x >> 5) + (threadIdx.x >> 5);
  unsigned lane = threadIdx.x & 31u;
  unsigned tnw  = wave & (NTN / 2 - 1u);          // tn-pair fastest: neighbors share A frags
  unsigned tmw  = wave >> 4;
  unsigned tm0  = tmw * 2u, tn0 = tnw * 2u;

  v8f acc[2][2][3] = {};                          // [tm sub][tn sub][gate]
#pragma unroll
  for (int kt = 0; kt < NTK; ++kt) {
    v16bf a0 = xa[((size_t)(tm0 + 0) * NTK + kt) * 32 + lane];
    v16bf a1 = xa[((size_t)(tm0 + 1) * NTK + kt) * 32 + lane];
#pragma unroll
    for (int g = 0; g < 3; ++g) {
      v16bf b0 = wB[g][((size_t)(tn0 + 0) * NTK + kt) * 32 + lane];
      acc[0][0][g] = __builtin_amdgcn_wmma_f32_16x16x32_bf16(false, a0, false, b0, (short)0, acc[0][0][g], false, false);
      acc[1][0][g] = __builtin_amdgcn_wmma_f32_16x16x32_bf16(false, a1, false, b0, (short)0, acc[1][0][g], false, false);
      v16bf b1 = wB[g][((size_t)(tn0 + 1) * NTK + kt) * 32 + lane];
      acc[0][1][g] = __builtin_amdgcn_wmma_f32_16x16x32_bf16(false, a0, false, b1, (short)0, acc[0][1][g], false, false);
      acc[1][1][g] = __builtin_amdgcn_wmma_f32_16x16x32_bf16(false, a1, false, b1, (short)0, acc[1][1][g], false, false);
    }
  }

  // C/D layout: VGPR v -> row M = v + (lane<16 ? 0 : 8), col N = lane%16
  unsigned col   = lane & 15u;
  unsigned rbase = (lane >> 4) * 8u;
#pragma unroll
  for (int j = 0; j < 2; ++j) {
    unsigned hg  = (tn0 + (unsigned)j) * 16u + col;
    float bfv = b_f[hg], biv = b_i[hg], bhv = b_h[hg];
#pragma unroll
    for (int i = 0; i < 2; ++i) {
      unsigned rb = (tm0 + (unsigned)i) * 16u + rbase;
#pragma unroll
      for (int v = 0; v < 8; ++v) {
        unsigned r  = rb + (unsigned)v;             // global row in [0, B*T)
        float fp = acc[i][j][0][v] + bfv;
        float ip = acc[i][j][1][v] + biv;
        float hp = acc[i][j][2][v] + bhv;
        float diff = softplus_f(-fp) - softplus_f(-ip);
        float lf = -softplus_f(diff);               // log forget gate
        float li = -softplus_f(-diff);              // log input gate
        float lv = li + log_g_f(hp);                // log_i + log_tilde_h
        size_t off = (size_t)r * H_ + hg;
        logF[off] = lf;
        logV[off] = lv;
      }
    }
  }
}

// ---------- kernel 4: per-chunk scan summaries ----------
// Composition over (A,B): running' = logaddexp(lf + running, lv)
// Chunk summary: A = sum(lf), B = chunk scan from identity (-inf)
__global__ void scan_chunk_kernel(const float* __restrict__ logF,
                                  const float* __restrict__ logV,
                                  float* __restrict__ sumA, float* __restrict__ sumB) {
  unsigned idx = blockIdx.x * blockDim.x + threadIdx.x;   // < B*CHK*H
  unsigned h = idx & (H_ - 1u);
  unsigned c = (idx >> 9) & (CHK - 1u);
  unsigned b = idx >> 15;
  float A = 0.f, Bv = -__builtin_inff();
  size_t base = ((size_t)b * T_ + (size_t)c * CL) * H_ + h;
  for (int i = 0; i < CL; ++i) {
    float lf = logF[base], lv = logV[base];
    A += lf;
    Bv = logaddexp_f(lf + Bv, lv);
    base += H_;
  }
  sumA[idx] = A;    // layout: (b*CHK + c)*H + h  == idx
  sumB[idx] = Bv;
}

// ---------- kernel 5: sequential carry scan across chunks (tiny) ----------
__global__ void scan_carry_kernel(const float* __restrict__ h0,
                                  const float* __restrict__ sumA,
                                  const float* __restrict__ sumB,
                                  float* __restrict__ pref, float* __restrict__ out) {
  unsigned idx = blockIdx.x * blockDim.x + threadIdx.x;   // < B*H
  unsigned h = idx & (H_ - 1u);
  unsigned b = idx >> 9;
  float run = log_g_f(h0[idx]);                           // h_0 is [B,1,H] flat
  out[((size_t)b * (T_ + 1)) * H_ + h] = __expf(run);     // slot t=0
  for (int c = 0; c < CHK; ++c) {
    size_t s = ((size_t)b * CHK + c) * H_ + h;
    pref[s] = run;
    run = logaddexp_f(sumA[s] + run, sumB[s]);
  }
}

// ---------- kernel 6: apply chunk prefixes, emit exp(log_h) ----------
__global__ void scan_apply_kernel(const float* __restrict__ logF,
                                  const float* __restrict__ logV,
                                  const float* __restrict__ pref,
                                  float* __restrict__ out) {
  unsigned idx = blockIdx.x * blockDim.x + threadIdx.x;   // < B*CHK*H
  unsigned h = idx & (H_ - 1u);
  unsigned c = (idx >> 9) & (CHK - 1u);
  unsigned b = idx >> 15;
  float run = pref[idx];
  size_t src = ((size_t)b * T_ + (size_t)c * CL) * H_ + h;
  size_t dst = ((size_t)b * (T_ + 1) + (size_t)c * CL + 1) * H_ + h;
  for (int i = 0; i < CL; ++i) {
    float lf = logF[src], lv = logV[src];
    run = logaddexp_f(lf + run, lv);
    out[dst] = __expf(run);
    src += H_; dst += H_;
  }
}

extern "C" void kernel_launch(void* const* d_in, const int* in_sizes, int n_in,
                              void* d_out, int out_size, void* d_ws, size_t ws_size,
                              hipStream_t stream) {
  (void)in_sizes; (void)n_in; (void)out_size; (void)ws_size;
  const float* x   = (const float*)d_in[0];
  const float* h0  = (const float*)d_in[1];
  const float* Wf  = (const float*)d_in[2];
  const float* bfv = (const float*)d_in[3];
  const float* Wi  = (const float*)d_in[4];
  const float* biv = (const float*)d_in[5];
  const float* Wh  = (const float*)d_in[6];
  const float* bhv = (const float*)d_in[7];
  float* out = (float*)d_out;

  char* ws = (char*)d_ws;
  size_t o = 0;
  uint32_t* xp   = (uint32_t*)(ws + o); o += (size_t)NTM * NTK * 32 * 8 * 4;      // 33.55 MB
  uint32_t* wp   = (uint32_t*)(ws + o); o += (size_t)3 * NTN * NTK * 32 * 8 * 4;  //  1.57 MB
  float*    logF = (float*)(ws + o);    o += (size_t)B_ * T_ * H_ * 4;            // 67.1 MB
  float*    logV = (float*)(ws + o);    o += (size_t)B_ * T_ * H_ * 4;            // 67.1 MB
  float*    sumA = (float*)(ws + o);    o += (size_t)B_ * CHK * H_ * 4;
  float*    sumB = (float*)(ws + o);    o += (size_t)B_ * CHK * H_ * 4;
  float*    pref = (float*)(ws + o);    o += (size_t)B_ * CHK * H_ * 4;

  pack_x_kernel<<<(NTM * NTK * 32 * 8) / 256, 256, 0, stream>>>(x, xp);
  pack_w_kernel<<<(3 * NTN * NTK * 32 * 8) / 256, 256, 0, stream>>>(Wf, Wi, Wh, wp);
  gemm_gates_kernel<<<(NTM / 2) * (NTN / 2) / 8, 256, 0, stream>>>(xp, wp, bfv, biv, bhv, logF, logV);
  scan_chunk_kernel<<<(B_ * CHK * H_) / 256, 256, 0, stream>>>(logF, logV, sumA, sumB);
  scan_carry_kernel<<<(B_ * H_) / 256, 256, 0, stream>>>(h0, sumA, sumB, pref, out);
  scan_apply_kernel<<<(B_ * CHK * H_) / 256, 256, 0, stream>>>(logF, logV, pref, out);
}
